// TransformerLayer_68418829025518
// MI455X (gfx1250) — matrix operand order repaired and checked by reading, not compiled, not run on hardware
//
#include <hip/hip_runtime.h>
#include <hip/hip_bf16.h>
#include <math.h>

// ---------------------------------------------------------------------------
// Problem constants (fixed by the reference)
// ---------------------------------------------------------------------------
constexpr int Bq   = 4;
constexpr int S    = 2047;
constexpr int T    = 2048;      // S + 1
constexpr int D    = 256;
constexpr int H    = 8;
constexpr int DK   = 32;
constexpr int Lnum = 2;
constexpr int DFF  = 512;
constexpr int M    = Bq * T;    // 8192 rows of activations
constexpr float EPSLN = 1e-5f;

typedef _Float16 v16h __attribute__((ext_vector_type(16)));
typedef _Float16 v8h  __attribute__((ext_vector_type(8)));
typedef float    v8f  __attribute__((ext_vector_type(8)));

__device__ inline v8f zero8() {
    v8f z;
#pragma unroll
    for (int i = 0; i < 8; ++i) z[i] = 0.0f;
    return z;
}

// ---------------------------------------------------------------------------
// Embedding: x[b,0,:] = embeddings[b,:]+pe[0,:]; x[b,t,:] = tok_emb[tok[b,t-1],:]+pe[t,:]
// Writes f32 x and f16 shadow xh.
// ---------------------------------------------------------------------------
__global__ void embed_kernel(const float* __restrict__ emb,
                             const int* __restrict__ tok,
                             const float* __restrict__ tok_emb,
                             const float* __restrict__ pe,
                             float* __restrict__ x,
                             _Float16* __restrict__ xh) {
    int idx = blockIdx.x * blockDim.x + threadIdx.x;     // < M*D = 2^21
    int d = idx & (D - 1);
    int t = (idx >> 8) & (T - 1);
    int b = idx >> 19;                                   // / (T*D)
    float v;
    if (t == 0) v = emb[b * D + d];
    else        v = tok_emb[(size_t)tok[b * S + (t - 1)] * D + d];
    v += pe[t * D + d];
    x[idx]  = v;
    xh[idx] = (_Float16)v;
}

// ---------------------------------------------------------------------------
// Weight convert + transpose: src f32 [K,N] row-major  ->  dst f16 [N,K]
// (B-operand wants contiguous K per output column)
// ---------------------------------------------------------------------------
__global__ void convT_kernel(const float* __restrict__ src,
                             _Float16* __restrict__ dst, int K, int N) {
    int idx = blockIdx.x * blockDim.x + threadIdx.x;
    if (idx >= K * N) return;
    int k = idx / N, n = idx % N;
    dst[(size_t)n * K + k] = (_Float16)src[idx];
}

// ---------------------------------------------------------------------------
// Generic WMMA GEMM:  C[M,N] = A[M,K](f16,row) x Bt[N,K](f16, pre-transposed)
// One wave computes a 16(M) x 64(N) tile; K consumed 32 at a time.
// Epilogues:
//  EPI=0: write f16 head-major q/k  [b,h,t,dk]
//  EPI=1: write f16 transposed v    [b,h,dk,t]
//  EPI=2: f32 out = acc + bias[n] + res[m,n]
//  EPI=3: f16 out = relu(acc + bias[n])
// ---------------------------------------------------------------------------
template <int EPI>
__global__ void gemm_wmma(const _Float16* __restrict__ A,
                          const _Float16* __restrict__ Bt,
                          const float* __restrict__ bias,
                          const float* __restrict__ res,
                          float* __restrict__ outF,
                          _Float16* __restrict__ outH,
                          int Mrows, int N, int K) {
    int wave = threadIdx.x >> 5, lane = threadIdx.x & 31;
    int gw = blockIdx.x * (blockDim.x >> 5) + wave;
    int tilesM = Mrows >> 4;
    int mt = gw % tilesM, nb = gw / tilesM;
    if (nb * 64 >= N) return;                            // wave-uniform
    int m0 = mt << 4, n0 = nb << 6;
    int hi = lane >> 4, lr = lane & 15;

    const _Float16* arow = A + (size_t)(m0 + lr) * K;
    v8f c0 = zero8(), c1 = zero8(), c2 = zero8(), c3 = zero8();

    for (int k0 = 0; k0 < K; k0 += 32) {
        v16h a;
        {   // A operand: row lr, chunks at K-offset 16*hi and 16*hi+8
            v8h lo = *(const v8h*)(arow + k0 + 16 * hi);
            v8h hp = *(const v8h*)(arow + k0 + 16 * hi + 8);
#pragma unroll
            for (int i = 0; i < 8; ++i) { a[i] = lo[i]; a[i + 8] = hp[i]; }
        }
        // B operand: column n, contiguous K run starting at 16*hi
        v16h b0 = *(const v16h*)(Bt + (size_t)(n0 +      lr) * K + k0 + 16 * hi);
        v16h b1 = *(const v16h*)(Bt + (size_t)(n0 + 16 + lr) * K + k0 + 16 * hi);
        v16h b2 = *(const v16h*)(Bt + (size_t)(n0 + 32 + lr) * K + k0 + 16 * hi);
        v16h b3 = *(const v16h*)(Bt + (size_t)(n0 + 48 + lr) * K + k0 + 16 * hi);
        c0 = __builtin_amdgcn_wmma_f32_16x16x32_f16(false, a, false, b0, (short)0, c0, false, false);
        c1 = __builtin_amdgcn_wmma_f32_16x16x32_f16(false, a, false, b1, (short)0, c1, false, false);
        c2 = __builtin_amdgcn_wmma_f32_16x16x32_f16(false, a, false, b2, (short)0, c2, false, false);
        c3 = __builtin_amdgcn_wmma_f32_16x16x32_f16(false, a, false, b3, (short)0, c3, false, false);
    }

    v8f cc[4] = {c0, c1, c2, c3};
#pragma unroll
    for (int nt = 0; nt < 4; ++nt) {
#pragma unroll
        for (int j = 0; j < 8; ++j) {
            int m = m0 + j + 8 * hi;
            int n = n0 + nt * 16 + lr;
            float v = cc[nt][j];
            if constexpr (EPI == 2) {
                v += bias[n] + res[(size_t)m * N + n];
                outF[(size_t)m * N + n] = v;
            } else if constexpr (EPI == 3) {
                v += bias[n];
                v = v > 0.0f ? v : 0.0f;
                outH[(size_t)m * N + n] = (_Float16)v;
            } else {
                int b = m >> 11, t = m & (T - 1);
                int h = n >> 5, dk = n & 31;
                if constexpr (EPI == 0)   // [b,h,t,dk]
                    outH[(((size_t)(b * H + h)) * T + t) * DK + dk] = (_Float16)v;
                else                      // [b,h,dk,t]
                    outH[(((size_t)(b * H + h)) * DK + dk) * T + t] = (_Float16)v;
            }
        }
    }
}

// ---------------------------------------------------------------------------
// Flash attention. One wave per (b,h, 16-query tile). Causal mask analytic.
// qf: [b,h,t,dk] f16 (A-layout rows), kf: [b,h,t,dk] f16 (B-layout for QK^T),
// vt: [b,h,dk,t] f16 (B-layout for PV). att out: [b,t, h*32+dk] f16.
// ---------------------------------------------------------------------------
__global__ void flash_attn(const _Float16* __restrict__ qf,
                           const _Float16* __restrict__ kf,
                           const _Float16* __restrict__ vt,
                           _Float16* __restrict__ att) {
    __shared__ __align__(32) _Float16 pst[4][16 * 32];   // 1KB per wave
    int wave = threadIdx.x >> 5, lane = threadIdx.x & 31;
    int gw = blockIdx.x * 4 + wave;                      // 0..4095
    int qt = gw & 127;                                   // T/16
    int bh = gw >> 7;                                    // 0..31
    int hi = lane >> 4, lr = lane & 15;
    _Float16* lds = pst[wave];

    // Q tile (A operand), loaded once
    const _Float16* qbase = qf + ((size_t)bh * T + qt * 16 + lr) * DK;
    v16h aQ;
    {
        v8h lo = *(const v8h*)(qbase + 16 * hi);
        v8h hp = *(const v8h*)(qbase + 16 * hi + 8);
#pragma unroll
        for (int i = 0; i < 8; ++i) { aQ[i] = lo[i]; aQ[i + 8] = hp[i]; }
    }

    float mrow[8], lrow[8];
#pragma unroll
    for (int j = 0; j < 8; ++j) { mrow[j] = -INFINITY; lrow[j] = 0.0f; }
    v8f acc0 = zero8(), acc1 = zero8();

    const float scl = 0.176776695296636893f;             // 1/sqrt(32)
    const _Float16* krow = kf + (size_t)bh * T * DK;
    const _Float16* vrow = vt + (size_t)bh * DK * T;
    int nkb = ((qt * 16 + 15) >> 5) + 1;                 // 32-key blocks

    for (int kb = 0; kb < nkb; ++kb) {
        int k0 = kb * 32;
        // scores: two 16x16 tiles (cols k0..k0+15, k0+16..k0+31)
        v16h bK0 = *(const v16h*)(krow + (size_t)(k0 +      lr) * DK + 16 * hi);
        v16h bK1 = *(const v16h*)(krow + (size_t)(k0 + 16 + lr) * DK + 16 * hi);
        v8f z = zero8();
        v8f s0 = __builtin_amdgcn_wmma_f32_16x16x32_f16(false, aQ, false, bK0, (short)0, z, false, false);
        v8f s1 = __builtin_amdgcn_wmma_f32_16x16x32_f16(false, aQ, false, bK1, (short)0, z, false, false);

        int c0 = k0 + lr, c1 = k0 + 16 + lr;
        float p0[8], p1[8];
#pragma unroll
        for (int j = 0; j < 8; ++j) {
            int qa = qt * 16 + j + 8 * hi;
            float v0 = (c0 <= qa) ? s0[j] * scl : -INFINITY;
            float v1 = (c1 <= qa) ? s1[j] * scl : -INFINITY;
            float mx = fmaxf(v0, v1);                    // row reduce over 16 lanes (half-wave)
            mx = fmaxf(mx, __shfl_xor(mx, 1, 32));
            mx = fmaxf(mx, __shfl_xor(mx, 2, 32));
            mx = fmaxf(mx, __shfl_xor(mx, 4, 32));
            mx = fmaxf(mx, __shfl_xor(mx, 8, 32));
            float mn = fmaxf(mrow[j], mx);
            float alpha = __expf(mrow[j] - mn);          // 0 on first block (m=-inf)
            mrow[j] = mn;
            p0[j] = __expf(v0 - mn);
            p1[j] = __expf(v1 - mn);
            float sm = p0[j] + p1[j];
            sm += __shfl_xor(sm, 1, 32);
            sm += __shfl_xor(sm, 2, 32);
            sm += __shfl_xor(sm, 4, 32);
            sm += __shfl_xor(sm, 8, 32);
            lrow[j] = lrow[j] * alpha + sm;
            acc0[j] *= alpha;
            acc1[j] *= alpha;
        }

        // repack P: C-layout f32 -> A-layout f16 via LDS (wave-private region)
#pragma unroll
        for (int j = 0; j < 8; ++j) {
            int r = j + 8 * hi;
            lds[r * 32 + lr]      = (_Float16)p0[j];
            lds[r * 32 + 16 + lr] = (_Float16)p1[j];
        }
        asm volatile("s_wait_dscnt 0" ::: "memory");
        v16h aP;
        {
            v8h lo = *(const v8h*)(lds + lr * 32 + 16 * hi);
            v8h hp = *(const v8h*)(lds + lr * 32 + 16 * hi + 8);
#pragma unroll
            for (int i = 0; i < 8; ++i) { aP[i] = lo[i]; aP[i + 8] = hp[i]; }
        }
        // V tiles (B operand, K = 32 keys of this block)
        v16h bV0 = *(const v16h*)(vrow + (size_t)lr * T        + k0 + 16 * hi);
        v16h bV1 = *(const v16h*)(vrow + (size_t)(16 + lr) * T + k0 + 16 * hi);
        acc0 = __builtin_amdgcn_wmma_f32_16x16x32_f16(false, aP, false, bV0, (short)0, acc0, false, false);
        acc1 = __builtin_amdgcn_wmma_f32_16x16x32_f16(false, aP, false, bV1, (short)0, acc1, false, false);
    }

    // normalize + write att [b, t, h*32+dk] f16
    int b = bh >> 3, h = bh & 7;
#pragma unroll
    for (int j = 0; j < 8; ++j) {
        int t = qt * 16 + j + 8 * hi;
        float inv = 1.0f / lrow[j];
        size_t base = ((size_t)(b * T + t)) * D + h * DK;
        att[base + lr]      = (_Float16)(acc0[j] * inv);
        att[base + 16 + lr] = (_Float16)(acc1[j] * inv);
    }
}

// ---------------------------------------------------------------------------
// LayerNorm over D=256: one wave per row (8 values/lane), full-wave reductions.
// Writes f32 x and f16 shadow xh.
// ---------------------------------------------------------------------------
__global__ void ln_kernel(const float* __restrict__ y,
                          const float* __restrict__ g,
                          const float* __restrict__ bta,
                          float* __restrict__ xout,
                          _Float16* __restrict__ xh) {
    int wave = threadIdx.x >> 5, lane = threadIdx.x & 31;
    int row = blockIdx.x * (blockDim.x >> 5) + wave;
    const float* yr = y + (size_t)row * D;
    float v[8];
    float4 u0 = *(const float4*)(yr + lane * 8);
    float4 u1 = *(const float4*)(yr + lane * 8 + 4);
    v[0] = u0.x; v[1] = u0.y; v[2] = u0.z; v[3] = u0.w;
    v[4] = u1.x; v[5] = u1.y; v[6] = u1.z; v[7] = u1.w;
    float s = 0.0f;
#pragma unroll
    for (int i = 0; i < 8; ++i) s += v[i];
    for (int m = 16; m >= 1; m >>= 1) s += __shfl_xor(s, m, 32);
    float mean = s * (1.0f / D);
    float q = 0.0f;
#pragma unroll
    for (int i = 0; i < 8; ++i) { float d = v[i] - mean; q += d * d; }
    for (int m = 16; m >= 1; m >>= 1) q += __shfl_xor(q, m, 32);
    float inv = rsqrtf(q * (1.0f / D) + EPSLN);
#pragma unroll
    for (int i = 0; i < 8; ++i) {
        int d = lane * 8 + i;
        float o = (v[i] - mean) * inv * g[d] + bta[d];
        xout[(size_t)row * D + d] = o;
        xh[(size_t)row * D + d]   = (_Float16)o;
    }
}

// ---------------------------------------------------------------------------
// Output: drop prepended row; out[b,s,d] = x[b,s+1,d]
// ---------------------------------------------------------------------------
__global__ void outcopy_kernel(const float* __restrict__ x, float* __restrict__ out) {
    int idx = blockIdx.x * blockDim.x + threadIdx.x;
    if (idx >= Bq * S * D) return;
    int d = idx & (D - 1);
    int r = idx >> 8;                    // b*S + s
    int b = r / S, s = r % S;
    out[idx] = x[((size_t)(b * T + s + 1)) * D + d];
}

// ---------------------------------------------------------------------------
// Launcher
// ---------------------------------------------------------------------------
extern "C" void kernel_launch(void* const* d_in, const int* in_sizes, int n_in,
                              void* d_out, int out_size, void* d_ws, size_t ws_size,
                              hipStream_t stream) {
    (void)in_sizes; (void)n_in; (void)out_size; (void)ws_size;
    const float* embeddings = (const float*)d_in[0];
    const int*   tokens     = (const int*)d_in[1];
    // d_in[2] causal_mask, d_in[3] key_padding_mask: applied analytically
    const float* tok_emb = (const float*)d_in[4];
    const float* pe      = (const float*)d_in[5];
    const float* Wq = (const float*)d_in[6];
    const float* Wk = (const float*)d_in[7];
    const float* Wv = (const float*)d_in[8];
    const float* Wo = (const float*)d_in[9];
    const float* bo = (const float*)d_in[10];
    const float* g1 = (const float*)d_in[11];
    const float* be1 = (const float*)d_in[12];
    const float* W1 = (const float*)d_in[13];
    const float* b1 = (const float*)d_in[14];
    const float* W2 = (const float*)d_in[15];
    const float* b2 = (const float*)d_in[16];
    const float* g2 = (const float*)d_in[17];
    const float* be2 = (const float*)d_in[18];
    float* out = (float*)d_out;

    // ---- workspace partition (≈47 MB) ----
    char* w = (char*)d_ws;
    auto alloc = [&](size_t bytes) -> void* {
        void* p = (void*)w;
        w += (bytes + 255) & ~(size_t)255;
        return p;
    };
    float*     x   = (float*)alloc((size_t)M * D * 4);
    _Float16*  xh  = (_Float16*)alloc((size_t)M * D * 2);
    float*     y   = (float*)alloc((size_t)M * D * 4);
    _Float16*  qf  = (_Float16*)alloc((size_t)Bq * H * T * DK * 2);
    _Float16*  kf  = (_Float16*)alloc((size_t)Bq * H * T * DK * 2);
    _Float16*  vtf = (_Float16*)alloc((size_t)Bq * H * T * DK * 2);
    _Float16*  att = (_Float16*)alloc((size_t)M * D * 2);
    _Float16*  h1  = (_Float16*)alloc((size_t)M * DFF * 2);
    _Float16*  wqT = (_Float16*)alloc((size_t)D * D * 2);
    _Float16*  wkT = (_Float16*)alloc((size_t)D * D * 2);
    _Float16*  wvT = (_Float16*)alloc((size_t)D * D * 2);
    _Float16*  woT = (_Float16*)alloc((size_t)D * D * 2);
    _Float16*  w1T = (_Float16*)alloc((size_t)D * DFF * 2);
    _Float16*  w2T = (_Float16*)alloc((size_t)DFF * D * 2);

    // ---- embeddings + positional encoding ----
    embed_kernel<<<(M * D) / 256, 256, 0, stream>>>(embeddings, tokens, tok_emb, pe, x, xh);

    for (int l = 0; l < Lnum; ++l) {
        const float* Wql = Wq + (size_t)l * D * D;
        const float* Wkl = Wk + (size_t)l * D * D;
        const float* Wvl = Wv + (size_t)l * D * D;
        const float* Wol = Wo + (size_t)l * D * D;
        const float* W1l = W1 + (size_t)l * D * DFF;
        const float* W2l = W2 + (size_t)l * DFF * D;

        // weight convert + transpose to f16 [N][K]
        convT_kernel<<<(D * D) / 256, 256, 0, stream>>>(Wql, wqT, D, D);
        convT_kernel<<<(D * D) / 256, 256, 0, stream>>>(Wkl, wkT, D, D);
        convT_kernel<<<(D * D) / 256, 256, 0, stream>>>(Wvl, wvT, D, D);
        convT_kernel<<<(D * D) / 256, 256, 0, stream>>>(Wol, woT, D, D);
        convT_kernel<<<(D * DFF) / 256, 256, 0, stream>>>(W1l, w1T, D, DFF);
        convT_kernel<<<(DFF * D) / 256, 256, 0, stream>>>(W2l, w2T, DFF, D);

        // QKV projections (M/16 * N/64 tiles, 4 waves/block)
        int gQ = ((M / 16) * (D / 64)) / 4;     // 512 blocks
        gemm_wmma<0><<<gQ, 128, 0, stream>>>(xh, wqT, nullptr, nullptr, nullptr, qf,  M, D, D);
        gemm_wmma<0><<<gQ, 128, 0, stream>>>(xh, wkT, nullptr, nullptr, nullptr, kf,  M, D, D);
        gemm_wmma<1><<<gQ, 128, 0, stream>>>(xh, wvT, nullptr, nullptr, nullptr, vtf, M, D, D);

        // flash attention: Bq*H*(T/16) = 4096 waves
        flash_attn<<<1024, 128, 0, stream>>>(qf, kf, vtf, att);

        // output projection + bias + residual -> y (f32)
        gemm_wmma<2><<<gQ, 128, 0, stream>>>(att, woT, bo + l * D, x, y, nullptr, M, D, D);
        ln_kernel<<<M / 4, 128, 0, stream>>>(y, g1 + l * D, be1 + l * D, x, xh);

        // FFN
        int gF1 = ((M / 16) * (DFF / 64)) / 4;  // 1024 blocks
        gemm_wmma<3><<<gF1, 128, 0, stream>>>(xh, w1T, b1 + l * DFF, nullptr, nullptr, h1, M, DFF, D);
        gemm_wmma<2><<<gQ, 128, 0, stream>>>(h1, w2T, b2 + l * D, x, y, nullptr, M, D, DFF);
        ln_kernel<<<M / 4, 128, 0, stream>>>(y, g2 + l * D, be2 + l * D, x, xh);
    }

    outcopy_kernel<<<(Bq * S * D + 255) / 256, 256, 0, stream>>>(x, out);
}
